// GraphSAGE_84593675862497
// MI455X (gfx1250) — matrix operand order repaired and checked by reading, not compiled, not run on hardware
//
#include <hip/hip_runtime.h>
#include <hip/hip_bf16.h>
#include <stdint.h>

#define N_NODES 50000
#define NP      50016   // padded to multiple of 16 for WMMA tiles
#define N_EDGES 800000
#define HID     128
#define BN_EPS  1e-5f

// ---------------- bf16 helpers / WMMA types ----------------
typedef __bf16 bf16_t;
typedef bf16_t v16bf __attribute__((ext_vector_type(16)));
typedef float  v8f   __attribute__((ext_vector_type(8)));
typedef unsigned int v4u __attribute__((ext_vector_type(4)));

__device__ __forceinline__ unsigned short f32_to_bf16(float f) {
  union { float f; uint32_t u; } v; v.f = f;
  uint32_t u = v.u;
  u += 0x7FFFu + ((u >> 16) & 1u);   // round-to-nearest-even
  return (unsigned short)(u >> 16);
}

// ---------------- utility kernels ----------------
__global__ void k_zero(float* __restrict__ p, int n) {
  int i = blockIdx.x * blockDim.x + threadIdx.x;
  if (i < n) p[i] = 0.0f;
}

__global__ void k_deg(const int* __restrict__ ei, float* __restrict__ deg) {
  int e = blockIdx.x * blockDim.x + threadIdx.x;
  if (e < N_EDGES) atomicAdd(&deg[ei[N_EDGES + e]], 1.0f);
}

__global__ void k_invdeg(const float* __restrict__ deg, float* __restrict__ inv) {
  int n = blockIdx.x * blockDim.x + threadIdx.x;
  if (n < NP) inv[n] = (n < N_NODES) ? (1.0f / fmaxf(deg[n], 1.0f)) : 0.0f;
}

// generic scalar linear: out[n][c] = relu?(b[c] + sum_k in[n][k]*W[c][k])
__global__ void k_lin(const float* __restrict__ in, const float* __restrict__ W,
                      const float* __restrict__ b, float* __restrict__ out,
                      int n, int din, int dout, int do_relu) {
  int idx = blockIdx.x * blockDim.x + threadIdx.x;
  if (idx >= n * dout) return;
  int node = idx / dout, c = idx % dout;
  const float* ip = in + (size_t)node * din;
  const float* wp = W  + (size_t)c * din;
  float acc = b[c];
  for (int k = 0; k < din; ++k) acc += ip[k] * wp[k];
  if (do_relu) acc = fmaxf(acc, 0.0f);
  out[(size_t)node * dout + c] = acc;
}

// per-edge scatter-add, float4 chunks (chunks = din/4)
__global__ void k_scatter_add(const float* __restrict__ z, const int* __restrict__ ei,
                              float* __restrict__ agg, int chunks, int din) {
  int idx = blockIdx.x * blockDim.x + threadIdx.x;
  int total = N_EDGES * chunks;
  if (idx >= total) return;
  int e = idx / chunks;
  int c = (idx - e * chunks) * 4;
  int src = ei[e];
  int dst = ei[N_EDGES + e];
  const float* zp = z   + (size_t)src * din + c;
  float*       ap = agg + (size_t)dst * din + c;
  atomicAdd(ap + 0, zp[0]);
  atomicAdd(ap + 1, zp[1]);
  atomicAdd(ap + 2, zp[2]);
  atomicAdd(ap + 3, zp[3]);
}

// build bf16 A = [agg*invdeg | z], padded rows -> 0
__global__ void k_pack_a(const float* __restrict__ agg, const float* __restrict__ inv,
                         const float* __restrict__ z, unsigned short* __restrict__ A,
                         int din) {
  int K = 2 * din;
  int idx = blockIdx.x * blockDim.x + threadIdx.x;
  if (idx >= NP * K) return;
  int n = idx / K, k = idx - n * K;
  float v = 0.0f;
  if (n < N_NODES)
    v = (k < din) ? agg[(size_t)n * din + k] * inv[n]
                  : z[(size_t)n * din + (k - din)];
  A[idx] = f32_to_bf16(v);
}

// pack Wcat[c][0..din)=Wl[c][:], [din..2din)=Wr[c][:] as bf16
__global__ void k_pack_w(const float* __restrict__ Wl, const float* __restrict__ Wr,
                         unsigned short* __restrict__ out, int dout, int din) {
  int K = 2 * din;
  int idx = blockIdx.x * blockDim.x + threadIdx.x;
  if (idx >= dout * K) return;
  int c = idx / K, k = idx - c * K;
  float v = (k < din) ? Wl[(size_t)c * din + k] : Wr[(size_t)c * din + (k - din)];
  out[idx] = f32_to_bf16(v);
}

// ---------------- WMMA GEMM: out = A(rowsPad x K) @ W(dout x K)^T + bias ----------------
// One wave per 16x16 output tile; bf16 inputs, f32 accumulate.
__global__ void k_wmma_gemm(const unsigned short* __restrict__ A,
                            const unsigned short* __restrict__ W,
                            const float* __restrict__ bias,
                            float* __restrict__ out,
                            int tilesM, int tilesN, int K, int dout) {
  int wave = (blockIdx.x * blockDim.x + threadIdx.x) >> 5;
  if (wave >= tilesM * tilesN) return;        // wave-uniform; EXEC stays all-ones
  int tm = wave / tilesN;
  int tn = wave - tm * tilesN;
  int lane = threadIdx.x & 31;
  int l15  = lane & 15;
  int hi   = lane >> 4;                       // 0: lanes 0-15, 1: lanes 16-31

  // A fragment (16x32 bf16, ISA 7.12.2): lane-half K offset 0/8, VGPRs 4-7 at +16
  const unsigned short* ap = A + (size_t)(tm * 16 + l15) * K + hi * 8;
  // B fragment (32x16 bf16): same K layout as A (lane-half K offset 0/8, VGPRs 4-7 at +16)
  const unsigned short* bp = W + (size_t)(tn * 16 + l15) * K + hi * 8;

  v8f acc = {0.f, 0.f, 0.f, 0.f, 0.f, 0.f, 0.f, 0.f};
  for (int k = 0; k < K; k += 32) {
    union { v16bf v; v4u q[2]; } a, b;
    a.q[0] = *(const v4u*)(ap + k);
    a.q[1] = *(const v4u*)(ap + k + 16);
    b.q[0] = *(const v4u*)(bp + k);
    b.q[1] = *(const v4u*)(bp + k + 16);
    acc = __builtin_amdgcn_wmma_f32_16x16x32_bf16(
        /*neg_a=*/false, a.v, /*neg_b=*/false, b.v,
        /*c_mod=*/(short)0, acc, /*reuse_a=*/false, /*reuse_b=*/false);
  }

  // C/D layout: VGPR j -> row tm*16 + j + hi*8, col tn*16 + l15
  int col  = tn * 16 + l15;
  float bc = bias[col];
  int row0 = tm * 16 + hi * 8;
#pragma unroll
  for (int j = 0; j < 8; ++j)
    out[(size_t)(row0 + j) * dout + col] = acc[j] + bc;
}

// ---------------- BatchNorm ----------------
#define BN_ROWS 256
__global__ void k_bn_stats(const float* __restrict__ z, float* __restrict__ stats) {
  int c  = threadIdx.x;                 // 128 threads = 128 channels
  int r0 = blockIdx.x * BN_ROWS;
  int r1 = r0 + BN_ROWS; if (r1 > N_NODES) r1 = N_NODES;
  float s = 0.f, s2 = 0.f;
  for (int r = r0; r < r1; ++r) {
    float v = z[(size_t)r * HID + c];
    s += v; s2 += v * v;
  }
  atomicAdd(&stats[c], s);
  atomicAdd(&stats[HID + c], s2);
}

__global__ void k_bn_apply(float* __restrict__ z, const float* __restrict__ stats,
                           const float* __restrict__ gamma, const float* __restrict__ beta) {
  int idx = blockIdx.x * blockDim.x + threadIdx.x;
  if (idx >= N_NODES * HID) return;
  int c = idx & (HID - 1);
  const float invN = 1.0f / (float)N_NODES;
  float m   = stats[c] * invN;
  float var = stats[HID + c] * invN - m * m;
  float v   = z[idx];
  z[idx] = fmaxf(gamma[c] * (v - m) * rsqrtf(var + BN_EPS) + beta[c], 0.0f);
}

// ---------------- host orchestration ----------------
static inline unsigned gs(long long n, int b) { return (unsigned)((n + b - 1) / b); }

extern "C" void kernel_launch(void* const* d_in, const int* in_sizes, int n_in,
                              void* d_out, int out_size, void* d_ws, size_t ws_size,
                              hipStream_t stream) {
  (void)in_sizes; (void)n_in; (void)out_size; (void)ws_size;

  const float* x     = (const float*)d_in[0];         // (N,7)
  const int*   ei    = (const int*)  d_in[1];         // (2,E)
  const float* encW1 = (const float*)d_in[2];
  const float* encb1 = (const float*)d_in[3];
  const float* encW2 = (const float*)d_in[4];
  const float* encb2 = (const float*)d_in[5];
  const float* sWl[4], *sbl[4], *sWr[4];
  for (int i = 0; i < 4; ++i) {
    sWl[i] = (const float*)d_in[6 + 3 * i];
    sbl[i] = (const float*)d_in[7 + 3 * i];
    sWr[i] = (const float*)d_in[8 + 3 * i];
  }
  const float* bng[3], *bnb[3];
  for (int i = 0; i < 3; ++i) {
    bng[i] = (const float*)d_in[18 + 2 * i];
    bnb[i] = (const float*)d_in[19 + 2 * i];
  }
  const float* decW1 = (const float*)d_in[24];
  const float* decb1 = (const float*)d_in[25];
  const float* decW2 = (const float*)d_in[26];
  const float* decb2 = (const float*)d_in[27];
  float* out = (float*)d_out;

  // workspace layout (256B-aligned slabs)
  char* ws = (char*)d_ws;
  size_t off = 0;
  auto alloc = [&](size_t bytes) -> void* {
    void* p = ws + off;
    off += (bytes + 255) & ~(size_t)255;
    return p;
  };
  float* Z0   = (float*)alloc((size_t)NP * HID * 4);
  float* Z1   = (float*)alloc((size_t)NP * HID * 4);
  float* Z2   = (float*)alloc((size_t)NP * HID * 4);
  float* AGG  = (float*)alloc((size_t)NP * HID * 4);
  unsigned short* ABUF = (unsigned short*)alloc((size_t)NP * 256 * 2);
  float* DEG  = (float*)alloc((size_t)NP * 4);
  float* INV  = (float*)alloc((size_t)NP * 4);
  unsigned short* WC[4];
  for (int i = 0; i < 4; ++i) WC[i] = (unsigned short*)alloc((size_t)128 * 256 * 2);
  float* STATS = (float*)alloc(256 * 4);

  const int B = 256;
  const int tilesM = NP / 16;  // 3126

  // degrees (shared by all convs)
  k_zero  <<<gs(NP, B), B, 0, stream>>>(DEG, NP);
  k_deg   <<<gs(N_EDGES, B), B, 0, stream>>>(ei, DEG);
  k_invdeg<<<gs(NP, B), B, 0, stream>>>(DEG, INV);

  // pack bf16 concatenated weights [Wl | Wr]
  k_pack_w<<<gs(128 * 128, B), B, 0, stream>>>(sWl[0], sWr[0], WC[0], 128, 64);
  k_pack_w<<<gs(128 * 256, B), B, 0, stream>>>(sWl[1], sWr[1], WC[1], 128, 128);
  k_pack_w<<<gs(128 * 256, B), B, 0, stream>>>(sWl[2], sWr[2], WC[2], 128, 128);
  k_pack_w<<<gs( 64 * 256, B), B, 0, stream>>>(sWl[3], sWr[3], WC[3],  64, 128);

  // encoder: 7 -> 64 (relu) -> 64
  k_lin<<<gs((long long)N_NODES * 64, B), B, 0, stream>>>(x,  encW1, encb1, Z0, N_NODES, 7,  64, 1);
  k_lin<<<gs((long long)N_NODES * 64, B), B, 0, stream>>>(Z0, encW2, encb2, Z1, N_NODES, 64, 64, 0);

  // ---- conv1: Z1(64) -> Z2(128), BN0+relu ----
  k_zero<<<gs(NP * 64, B), B, 0, stream>>>(AGG, NP * 64);
  k_scatter_add<<<gs((long long)N_EDGES * 16, B), B, 0, stream>>>(Z1, ei, AGG, 16, 64);
  k_pack_a<<<gs((long long)NP * 128, B), B, 0, stream>>>(AGG, INV, Z1, ABUF, 64);
  k_wmma_gemm<<<gs((long long)tilesM * 8 * 32, B), B, 0, stream>>>(ABUF, WC[0], sbl[0], Z2, tilesM, 8, 128, 128);
  k_zero<<<1, 256, 0, stream>>>(STATS, 256);
  k_bn_stats<<<gs(N_NODES, BN_ROWS), HID, 0, stream>>>(Z2, STATS);
  k_bn_apply<<<gs((long long)N_NODES * HID, B), B, 0, stream>>>(Z2, STATS, bng[0], bnb[0]);

  // ---- conv2: Z2(128) -> Z0(128), BN1+relu ----
  k_zero<<<gs(NP * HID, B), B, 0, stream>>>(AGG, NP * HID);
  k_scatter_add<<<gs((long long)N_EDGES * 32, B), B, 0, stream>>>(Z2, ei, AGG, 32, 128);
  k_pack_a<<<gs((long long)NP * 256, B), B, 0, stream>>>(AGG, INV, Z2, ABUF, 128);
  k_wmma_gemm<<<gs((long long)tilesM * 8 * 32, B), B, 0, stream>>>(ABUF, WC[1], sbl[1], Z0, tilesM, 8, 256, 128);
  k_zero<<<1, 256, 0, stream>>>(STATS, 256);
  k_bn_stats<<<gs(N_NODES, BN_ROWS), HID, 0, stream>>>(Z0, STATS);
  k_bn_apply<<<gs((long long)N_NODES * HID, B), B, 0, stream>>>(Z0, STATS, bng[1], bnb[1]);

  // ---- conv3: Z0(128) -> Z1(128), BN2+relu ----
  k_zero<<<gs(NP * HID, B), B, 0, stream>>>(AGG, NP * HID);
  k_scatter_add<<<gs((long long)N_EDGES * 32, B), B, 0, stream>>>(Z0, ei, AGG, 32, 128);
  k_pack_a<<<gs((long long)NP * 256, B), B, 0, stream>>>(AGG, INV, Z0, ABUF, 128);
  k_wmma_gemm<<<gs((long long)tilesM * 8 * 32, B), B, 0, stream>>>(ABUF, WC[2], sbl[2], Z1, tilesM, 8, 256, 128);
  k_zero<<<1, 256, 0, stream>>>(STATS, 256);
  k_bn_stats<<<gs(N_NODES, BN_ROWS), HID, 0, stream>>>(Z1, STATS);
  k_bn_apply<<<gs((long long)N_NODES * HID, B), B, 0, stream>>>(Z1, STATS, bng[2], bnb[2]);

  // ---- conv4: Z1(128) -> Z2(64), no BN ----
  k_zero<<<gs(NP * HID, B), B, 0, stream>>>(AGG, NP * HID);
  k_scatter_add<<<gs((long long)N_EDGES * 32, B), B, 0, stream>>>(Z1, ei, AGG, 32, 128);
  k_pack_a<<<gs((long long)NP * 256, B), B, 0, stream>>>(AGG, INV, Z1, ABUF, 128);
  k_wmma_gemm<<<gs((long long)tilesM * 4 * 32, B), B, 0, stream>>>(ABUF, WC[3], sbl[3], Z2, tilesM, 4, 256, 64);

  // decoder: 64 -> 64 (relu) -> 4
  k_lin<<<gs((long long)N_NODES * 64, B), B, 0, stream>>>(Z2, decW1, decb1, Z0, N_NODES, 64, 64, 1);
  k_lin<<<gs((long long)N_NODES * 4,  B), B, 0, stream>>>(Z0, decW2, decb2, out, N_NODES, 64, 4, 0);
}
